// LatentOddityQuantizer_33243046871475
// MI455X (gfx1250) — matrix operand order repaired and checked by reading, not compiled, not run on hardware
//
#include <hip/hip_runtime.h>
#include <hip/hip_bf16.h>

typedef __attribute__((ext_vector_type(16))) _Float16 v16h;
typedef __attribute__((ext_vector_type(8)))  _Float16 v8h;
typedef __attribute__((ext_vector_type(8)))  float    v8f;
typedef __attribute__((ext_vector_type(4)))  float    v4f;
typedef __attribute__((ext_vector_type(4)))  unsigned int u32x4;
typedef __attribute__((ext_vector_type(8)))  int      i32x8;
typedef __attribute__((ext_vector_type(4)))  int      i32x4;

#define NPOS   2048      // B*T
#define DDIM   64
#define KCODES 1024
#define KDIM   4224      // 4096 (G flat) + 64 (-2w) + 1 (c) padded to 33*128
#define MC     128       // m-chunk staged in LDS per iteration
#define GAMMA_F 10.0f

#if __has_builtin(__builtin_amdgcn_tensor_load_to_lds)
#define HAVE_TDM 1
#else
#define HAVE_TDM 0
#endif

// ---------------------------------------------------------------------------
// Kernel 1: w_n = G_n z_n  (f32, D=64),  c_n = z_n . w_n
// ---------------------------------------------------------------------------
__global__ void prep_w_kernel(const float* __restrict__ z,
                              const float* __restrict__ G,
                              float* __restrict__ w,
                              float* __restrict__ c) {
    const int n = blockIdx.x;
    const int j = threadIdx.x;            // 0..63
    __shared__ float zs[DDIM];
    __shared__ float prod[DDIM];
    zs[j] = z[n * DDIM + j];
    __syncthreads();
    const float* grow = G + (size_t)n * DDIM * DDIM + (size_t)j * DDIM;
    float acc = 0.0f;
#pragma unroll
    for (int e = 0; e < DDIM; e += 4) {
        v4f g = *(const v4f*)(grow + e);
        acc += g[0] * zs[e] + g[1] * zs[e + 1] + g[2] * zs[e + 2] + g[3] * zs[e + 3];
    }
    w[n * DDIM + j] = acc;
    prod[j] = acc * zs[j];
    __syncthreads();
    if (j == 0) {
        float s = 0.0f;
#pragma unroll
        for (int t = 0; t < DDIM; ++t) s += prod[t];
        c[n] = s;
    }
}

// ---------------------------------------------------------------------------
// Kernel 2: A16[n, m] (f16) = [ G_n flat | -2*w_n | c_n | zeros ]
// ---------------------------------------------------------------------------
__global__ void build_a16_kernel(const float* __restrict__ G,
                                 const float* __restrict__ w,
                                 const float* __restrict__ c,
                                 _Float16* __restrict__ A16) {
    const int n = blockIdx.x;
    const int tid = threadIdx.x;
    _Float16* row = A16 + (size_t)n * KDIM;
    const float* g = G + (size_t)n * 4096;
    for (int idx = tid; idx < KDIM; idx += 256) {
        float v;
        if (idx < 4096)       v = g[idx];
        else if (idx < 4160)  v = -2.0f * w[n * DDIM + (idx - 4096)];
        else if (idx == 4160) v = c[n];
        else                  v = 0.0f;
        row[idx] = (_Float16)v;
    }
}

// ---------------------------------------------------------------------------
// Kernel 3: P[k, m] (f16):  m<4096 -> e_k[m/64]*e_k[m%64]
//                           m<4160 -> e_k[m-4096]; m=4160 -> 1.0; else 0
// ---------------------------------------------------------------------------
__global__ void build_p_kernel(const float* __restrict__ emb,
                               _Float16* __restrict__ P) {
    const int k = blockIdx.x;
    const int tid = threadIdx.x;
    __shared__ float es[DDIM];
    if (tid < DDIM) es[tid] = emb[k * DDIM + tid];
    __syncthreads();
    _Float16* row = P + (size_t)k * KDIM;
    for (int idx = tid; idx < KDIM; idx += 256) {
        float v;
        if (idx < 4096)       v = es[idx >> 6] * es[idx & 63];
        else if (idx < 4160)  v = es[idx - 4096];
        else if (idx == 4160) v = 1.0f;
        else                  v = 0.0f;
        row[idx] = (_Float16)v;
    }
}

// ---------------------------------------------------------------------------
// TDM helpers: 2D tile load (tile_dim0=MC elems, tile_dim1=32 rows,
// stride=KDIM elems, data_size=2B) from A16 into LDS.  D# per ISA ch.8.
// ---------------------------------------------------------------------------
__device__ __forceinline__ void tdm_wait0() {
#if __has_builtin(__builtin_amdgcn_s_wait_tensorcnt)
    __builtin_amdgcn_s_wait_tensorcnt(0);
#else
    asm volatile("s_wait_tensorcnt 0x0" ::: "memory");
#endif
}

#if HAVE_TDM
__device__ __forceinline__ void tdm_load_tile(unsigned ldsOff,
                                              const _Float16* gptr) {
    unsigned long long ga = (unsigned long long)(size_t)gptr;
    u32x4 g0;
    g0[0] = 1u;                                     // count=1, user mode
    g0[1] = ldsOff;                                 // lds_addr (bytes)
    g0[2] = (unsigned)ga;                           // global_addr[31:0]
    g0[3] = (unsigned)((ga >> 32) & 0x01FFFFFFu)    // global_addr[56:32]
            | (2u << 30);                           // type=2 ("image")
    i32x8 g1;
    g1[0] = 1 << 16;                                // data_size=1 (2 bytes)
    g1[1] = (int)((KDIM & 0xFFFF) << 16);           // tensor_dim0[15:0]
    g1[2] = (int)((KDIM >> 16) & 0xFFFF)            // tensor_dim0[31:16]
            | (int)((NPOS & 0xFFFF) << 16);         // tensor_dim1[15:0]
    g1[3] = (int)((NPOS >> 16) & 0xFFFF)            // tensor_dim1[31:16]
            | (MC << 16);                           // tile_dim0 = MC
    g1[4] = 32;                                     // tile_dim1=32, tile_dim2=0
    g1[5] = KDIM;                                   // tensor_dim0_stride[31:0]
    g1[6] = 0;                                      // stride[47:32] | dim1_stride lo
    g1[7] = 0;
    i32x4 g2 = {0, 0, 0, 0};
    i32x4 g3 = {0, 0, 0, 0};
#if __clang_major__ >= 23
    i32x8 g4 = {0, 0, 0, 0, 0, 0, 0, 0};
    __builtin_amdgcn_tensor_load_to_lds(g0, g1, g2, g3, g4, 0);
#else
    __builtin_amdgcn_tensor_load_to_lds(g0, g1, g2, g3, 0);
#endif
}
#endif

// ---------------------------------------------------------------------------
// Kernel 4: sim[n,k] = exp(-gamma * (A16 @ P^T)[n,k]) via WMMA f16->f32
// Block: 256 thr (8 waves). Block tile: 32 rows x 256 cols.
// Wave tile: 32x32 (2 row-tiles x 2 k-tiles, B fragment reused across rows).
// A tile staged to LDS by the Tensor Data Mover, consumed by all 8 waves.
// ---------------------------------------------------------------------------
__global__ __launch_bounds__(256) void dist_wmma_kernel(
    const _Float16* __restrict__ A16, const _Float16* __restrict__ P,
    float* __restrict__ sim) {
    __shared__ __align__(32) _Float16 As[32][MC];

    const int tid  = threadIdx.x;
    const int lane = tid & 31;
    const int wave = tid >> 5;
    const int n0   = blockIdx.x * 32;
    const int k0   = blockIdx.y * 256 + wave * 32;

    v8f acc[2][2] = {};

    const int arow  = lane & 15;             // A fragment: M = lane%16
    const int ksel  = (lane >> 4) << 4;      // A K-base: 0 or 16 (contiguous 16)
    const int bbase = (lane >> 4) << 4;      // B K-base: 0 or 16
    const int bcol  = lane & 15;             // B fragment: N col = lane%16
#if HAVE_TDM
    const unsigned ldsBase = (unsigned)(size_t)&As[0][0];
#endif

    for (int mc = 0; mc < KDIM; mc += MC) {
#if HAVE_TDM
        if (wave == 0) {
            tdm_load_tile(ldsBase, A16 + (size_t)n0 * KDIM + mc);
            tdm_wait0();
        }
        __syncthreads();
#else
        // Fallback: cooperative b128 copy (2 x v8h per thread)
#pragma unroll
        for (int i = 0; i < 2; ++i) {
            const int idx  = tid + i * 256;       // v8h group id, 0..511
            const int row  = idx >> 4;            // MC/8 = 16 groups per row
            const int col8 = (idx & 15) << 3;
            *(v8h*)&As[row][col8] =
                *(const v8h*)(A16 + (size_t)(n0 + row) * KDIM + mc + col8);
        }
        __syncthreads();
#endif

#pragma unroll
        for (int ks = 0; ks < MC; ks += 32) {
            v16h a[2];
#pragma unroll
            for (int rt = 0; rt < 2; ++rt) {
                v8h lo = *(const v8h*)&As[rt * 16 + arow][ks + ksel];
                v8h hi = *(const v8h*)&As[rt * 16 + arow][ks + ksel + 8];
#pragma unroll
                for (int i = 0; i < 8; ++i) { a[rt][i] = lo[i]; a[rt][i + 8] = hi[i]; }
            }
#pragma unroll
            for (int t = 0; t < 2; ++t) {
                const _Float16* bp =
                    P + (size_t)(k0 + t * 16 + bcol) * KDIM + (mc + ks) + bbase;
                v16h b = *(const v16h*)bp;
#pragma unroll
                for (int rt = 0; rt < 2; ++rt) {
                    acc[rt][t] = __builtin_amdgcn_wmma_f32_16x16x32_f16(
                        false, a[rt], false, b, (short)0, acc[rt][t], false, false);
                }
            }
        }
        __syncthreads();
    }

    // ---- epilogue: sim = exp(-gamma * dist)
    const int rbase = (lane >> 4) << 3;   // D rows: r + 0/8
    const int ncol  = lane & 15;
#pragma unroll
    for (int rt = 0; rt < 2; ++rt) {
#pragma unroll
        for (int t = 0; t < 2; ++t) {
            const int kk = k0 + t * 16 + ncol;
#pragma unroll
            for (int r = 0; r < 8; ++r) {
                const int n = n0 + rt * 16 + rbase + r;
                sim[(size_t)n * KCODES + kk] = __expf(-GAMMA_F * acc[rt][t][r]);
            }
        }
    }
}

// ---------------------------------------------------------------------------
// Kernel 5: per-row argmax over K=1024 (one wave32 per row), first-max ties
// ---------------------------------------------------------------------------
__global__ void argmax_kernel(const float* __restrict__ sim,
                              int* __restrict__ idx_out) {
    const int n = blockIdx.x * 8 + (threadIdx.x >> 5);
    const int lane = threadIdx.x & 31;
    const float* row = sim + (size_t)n * KCODES;
    float bv = -1.0f;
    int bi = 0;
    for (int k = lane; k < KCODES; k += 32) {
        float v = row[k];
        if (v > bv) { bv = v; bi = k; }
    }
#pragma unroll
    for (int off = 16; off > 0; off >>= 1) {
        float ov = __shfl_xor(bv, off, 32);
        int   oi = __shfl_xor(bi, off, 32);
        if (ov > bv || (ov == bv && oi < bi)) { bv = ov; bi = oi; }
    }
    if (lane == 0) idx_out[n] = bi;
}

// ---------------------------------------------------------------------------
extern "C" void kernel_launch(void* const* d_in, const int* in_sizes, int n_in,
                              void* d_out, int out_size, void* d_ws, size_t ws_size,
                              hipStream_t stream) {
    const float* z   = (const float*)d_in[0];  // (B,T,D)
    const float* G   = (const float*)d_in[1];  // (N,D,D)
    const float* emb = (const float*)d_in[2];  // (K,D)

    int*   idx_out = (int*)d_out;              // first N int32 indices
    float* sim     = (float*)d_out + NPOS;     // then N*K f32 similarity

    char* ws = (char*)d_ws;
    size_t off = 0;
    _Float16* P   = (_Float16*)(ws + off); off += (size_t)KCODES * KDIM * 2;
    _Float16* A16 = (_Float16*)(ws + off); off += (size_t)NPOS * KDIM * 2;
    float*    w   = (float*)(ws + off);    off += (size_t)NPOS * DDIM * 4;
    float*    c   = (float*)(ws + off);

    prep_w_kernel<<<NPOS, 64, 0, stream>>>(z, G, w, c);
    build_a16_kernel<<<NPOS, 256, 0, stream>>>(G, w, c, A16);
    build_p_kernel<<<KCODES, 256, 0, stream>>>(emb, P);
    dim3 grid(NPOS / 32, KCODES / 256);
    dist_wmma_kernel<<<grid, 256, 0, stream>>>(A16, P, sim);
    argmax_kernel<<<NPOS / 8, 256, 0, stream>>>(sim, idx_out);
}
